// MultiHeadAttention_48507360641456
// MI455X (gfx1250) — matrix operand order repaired and checked by reading, not compiled, not run on hardware
//
#include <hip/hip_runtime.h>
#include <hip/hip_bf16.h>

typedef __attribute__((ext_vector_type(16))) _Float16 v16h;
typedef __attribute__((ext_vector_type(8)))  float    v8f;
typedef __attribute__((ext_vector_type(4)))  unsigned int u32x4;
typedef __attribute__((ext_vector_type(8)))  int      i32x8;
typedef __attribute__((ext_vector_type(4)))  int      i32x4;

#define N_SEQ 1024
#define BATCH 8
#define EMB   256
#define HEADS 8
#define DH    32
#define NBH   64
#define QKV_COLS 768
#define ROWS  (N_SEQ * BATCH)          // 8192 GEMM rows (n*B + b)

#if defined(__HIP_DEVICE_COMPILE__) && __has_builtin(__builtin_amdgcn_tensor_load_to_lds)
#define USE_TDM 1
#else
#define USE_TDM 0
#endif

// ---------------- WMMA helpers (gfx1250, wave32) ----------------

__device__ __forceinline__ v8f wmma_f16(v16h a, v16h b, v8f c) {
  // (neg_a, A, neg_b, B, c_mod, C, reuse_a, reuse_b)
  return __builtin_amdgcn_wmma_f32_16x16x32_f16(false, a, false, b, (short)0, c, false, false);
}

// A fragment 16x32 (MxK), f16 source, row stride ld (halves).
// lanes 0-15: row=lane, K={0..7,16..23}; lanes 16-31: row=lane-16, K={8..15,24..31}
__device__ __forceinline__ v16h load_a_f16(const _Float16* __restrict__ src, int ld) {
  int lane = threadIdx.x & 31;
  const _Float16* p = src + (lane & 15) * ld;
  int kof = (lane < 16) ? 0 : 8;
  v16h a;
#pragma unroll
  for (int i = 0; i < 8; ++i) { a[i] = p[kof + i]; a[8 + i] = p[16 + kof + i]; }
  return a;
}

__device__ __forceinline__ v16h load_a_f32(const float* __restrict__ src, int ld) {
  int lane = threadIdx.x & 31;
  const float* p = src + (lane & 15) * ld;
  int kof = (lane < 16) ? 0 : 8;
  v16h a;
#pragma unroll
  for (int i = 0; i < 8; ++i) { a[i] = (_Float16)p[kof + i]; a[8 + i] = (_Float16)p[16 + kof + i]; }
  return a;
}

// B fragment 32x16 (KxN) where B[k,n] = src[n*ld + k] (each n-row holds contiguous K).
// lanes 0-15: col=lane, K={0..7,16..23}; lanes 16-31: col=lane-16, K={8..15,24..31}
__device__ __forceinline__ v16h load_b_rows_f16(const _Float16* __restrict__ src, int ld) {
  int lane = threadIdx.x & 31;
  const _Float16* p = src + (lane & 15) * ld + ((lane < 16) ? 0 : 8);
  v16h b;
#pragma unroll
  for (int i = 0; i < 8; ++i) { b[i] = p[i]; b[8 + i] = p[16 + i]; }
  return b;
}

__device__ __forceinline__ v16h load_b_rows_f32(const float* __restrict__ src, int ld) {
  int lane = threadIdx.x & 31;
  const float* p = src + (lane & 15) * ld + ((lane < 16) ? 0 : 8);
  v16h b;
#pragma unroll
  for (int i = 0; i < 8; ++i) { b[i] = (_Float16)p[i]; b[8 + i] = (_Float16)p[16 + i]; }
  return b;
}

// ---------------- kernel 1: QKV projection ----------------
__global__ void qkv_proj_kernel(const float* __restrict__ value,
                                const float* __restrict__ w,
                                const float* __restrict__ bias,
                                _Float16* __restrict__ qh,
                                _Float16* __restrict__ kh,
                                _Float16* __restrict__ vt) {
  const int wave = threadIdx.x >> 5;
  const int lane = threadIdx.x & 31;
  const int tile = blockIdx.x * 8 + wave;
  const int tiles_j = QKV_COLS / 16;                 // 48
  const int mbase = (tile / tiles_j) * 16;
  const int jbase = (tile % tiles_j) * 16;

  v8f acc = {};
#pragma unroll
  for (int kb = 0; kb < EMB; kb += 32) {
    v16h a = load_a_f32(value + (size_t)mbase * EMB + kb, EMB);
    v16h b = load_b_rows_f32(w + (size_t)jbase * EMB + kb, EMB);
    acc = wmma_f16(a, b, acc);
  }

  const float scale = 0.17677669529663687f;          // 1/sqrt(Dh)
  const int cn = lane & 15;
  const int rofs = (lane >= 16) ? 8 : 0;
  const int j = jbase + cn;
  const float bj = bias[j];
#pragma unroll
  for (int i = 0; i < 8; ++i) {
    int row = mbase + i + rofs;
    int n = row >> 3;                                // row = n*8 + b
    int bi = row & 7;
    float vacc = acc[i] + bj;
    if (j < EMB) {
      int hh = j >> 5, dh = j & 31;
      qh[(((size_t)bi * HEADS + hh) * N_SEQ + n) * DH + dh] = (_Float16)(vacc * scale);
    } else if (j < 2 * EMB) {
      int jj = j - EMB; int hh = jj >> 5, dh = jj & 31;
      kh[(((size_t)bi * HEADS + hh) * N_SEQ + n) * DH + dh] = (_Float16)vacc;
    } else {
      int jj = j - 2 * EMB; int hh = jj >> 5, dh = jj & 31;
      vt[(((size_t)bi * HEADS + hh) * DH + dh) * N_SEQ + n] = (_Float16)vacc;  // transposed
    }
  }
}

// ---------------- kernel 2: fused gated attention ----------------
// TDM pads 1 dword every 256 dwords -> row stride 1028, element (r,c) at r*1028 + c + (c>>8)
#define SC_LD 1028
#define SC_FLOATS   (16 * SC_LD)
#define AVG_FLOATS  (16 * 1024)
#define PART_FLOATS (8 * 16 * 32)
#define RED_FLOATS  (16 * 16)
// layout is IDENTICAL with or without TDM so host launch size always matches device view
#define SMEM_FLOATS (2 * SC_FLOATS + AVG_FLOATS + PART_FLOATS + RED_FLOATS + 16)

__device__ __forceinline__ int scofs(int r, int c) { return r * SC_LD + c + (c >> 8); }

#if USE_TDM
// Issue a TDM load of a 16 x 1024 f32 tile (row stride N_SEQ) into LDS at lds_off,
// with 1-dword pad every 256 dwords (D# group layouts per CDNA5 ISA).
__device__ __forceinline__ void tdm_load_pos(const float* gsrc, unsigned lds_off) {
  unsigned long long ga = (unsigned long long)(size_t)gsrc;
  u32x4 g0;
  g0[0] = 1u;                                              // count=1 (valid user descriptor)
  g0[1] = lds_off;                                         // lds_addr (bytes)
  g0[2] = (unsigned)(ga & 0xffffffffu);                    // global_addr[31:0]
  g0[3] = (unsigned)((ga >> 32) & 0x01ffffffu) | (2u << 30); // global_addr[56:32] | type=2
  i32x8 g1;
  g1[0] = (int)((2u << 16) | (1u << 20) | (7u << 22));     // data_size=4B, pad_en, pad every 256dw, pad 1dw
  g1[1] = (int)(1024u << 16);                              // atomic_barrier=0 | tensor_dim0.lo=1024
  g1[2] = (int)(16u << 16);                                // tensor_dim0.hi=0 | tensor_dim1.lo=16
  g1[3] = (int)(1024u << 16);                              // tensor_dim1.hi=0 | tile_dim0=1024
  g1[4] = (int)16u;                                        // tile_dim1=16, tile_dim2=0
  g1[5] = (int)1024;                                       // tensor_dim0_stride[31:0]
  g1[6] = 0;                                               // stride hi, dim1_stride lo
  g1[7] = 0;
  i32x4 z4 = {};
#if __clang_major__ >= 23
  i32x8 z8 = {};
  __builtin_amdgcn_tensor_load_to_lds(g0, g1, z4, z4, z8, 0);
#else
  __builtin_amdgcn_tensor_load_to_lds(g0, g1, z4, z4, 0);
#endif
}
#endif

__global__ void attn_kernel(const _Float16* __restrict__ qh,
                            const _Float16* __restrict__ kh,
                            const _Float16* __restrict__ vt,
                            const float* __restrict__ pos,
                            float* __restrict__ ctx,
                            float* __restrict__ avg_out) {
  extern __shared__ float smem[];
  float* buf0 = smem;                        // 16 x SC_LD  (pos tile -> scores -> probs)
  float* buf1 = smem + SC_FLOATS;            // double buffer for TDM overlap
  float* avg  = smem + 2 * SC_FLOATS;        // 16 x 1024 head-averaged probs
  float* part = avg + AVG_FLOATS;            // 8 waves x 16 x 32 AV partials
  float* red  = part + PART_FLOATS;          // 16 x 16 reduction scratch
  float* rowstat = red + RED_FLOATS;         // 16

  const int b = blockIdx.x & 7;
  const int nbase = (blockIdx.x >> 3) * 16;
  const int wave = threadIdx.x >> 5;
  const int lane = threadIdx.x & 31;
  const int cn = lane & 15;
  const int rofs = (lane >= 16) ? 8 : 0;
  const int r16 = threadIdx.x >> 4;          // 0..15 (softmax row)
  const int t16 = threadIdx.x & 15;          // 0..15 (softmax col group)

  const float* posbase = pos + (size_t)(b * HEADS) * N_SEQ * N_SEQ + (size_t)nbase * N_SEQ;

  for (int e = threadIdx.x; e < AVG_FLOATS; e += 256) avg[e] = 0.0f;
  __syncthreads();

#if USE_TDM
  if (wave == 0) tdm_load_pos(posbase, (unsigned)(size_t)(void*)buf0);
#endif

  for (int h = 0; h < HEADS; ++h) {
    const int bh = b * HEADS + h;
    const _Float16* qhead  = qh + (size_t)bh * N_SEQ * DH;
    const _Float16* khead  = kh + (size_t)bh * N_SEQ * DH;
    const _Float16* vthead = vt + (size_t)bh * DH * N_SEQ;
    float* cur = (h & 1) ? buf1 : buf0;

#if USE_TDM
    if (wave == 0) {
      if (h < HEADS - 1) {
        // prefetch next head's pos tile into the other buffer, then wait for current
        tdm_load_pos(posbase + (size_t)(h + 1) * N_SEQ * N_SEQ,
                     (unsigned)(size_t)(void*)((h & 1) ? buf0 : buf1));
        __builtin_amdgcn_s_wait_tensorcnt(1);
      } else {
        __builtin_amdgcn_s_wait_tensorcnt(0);
      }
    }
    __syncthreads();                            // pos tile for head h visible to all waves
#else
    const float* phead = posbase + (size_t)h * N_SEQ * N_SEQ;
#endif

    // ---- content scores + log-sigmoid gate: wave w covers cols [w*128, w*128+128) ----
    v16h aq = load_a_f16(qhead + (size_t)nbase * DH, DH);
#pragma unroll
    for (int t = 0; t < 8; ++t) {
      const int col = wave * 128 + t * 16;
      v16h bk = load_b_rows_f16(khead + (size_t)col * DH, DH);
      v8f c = {};
      c = wmma_f16(aq, bk, c);
#pragma unroll
      for (int i = 0; i < 8; ++i) {
        int r = i + rofs;
        float s = fminf(fmaxf(c[i], -10.0f), 10.0f);
#if USE_TDM
        float p = cur[scofs(r, col + cn)];
#else
        float p = phead[(size_t)r * N_SEQ + col + cn];
#endif
        p = fminf(fmaxf(p, -10.0f), 10.0f);
        s -= __logf(1.0f + __expf(-p));             // + log_sigmoid(pos)
        cur[scofs(r, col + cn)] = s;
      }
    }
    __syncthreads();

    // ---- softmax over the 1024-wide row ----
    float m = -1e30f;
    for (int c0 = t16; c0 < 1024; c0 += 16) m = fmaxf(m, cur[scofs(r16, c0)]);
    red[r16 * 16 + t16] = m;
    __syncthreads();
    if (t16 == 0) {
      float mm = red[r16 * 16];
#pragma unroll
      for (int i = 1; i < 16; ++i) mm = fmaxf(mm, red[r16 * 16 + i]);
      rowstat[r16] = mm;
    }
    __syncthreads();
    const float mm = rowstat[r16];
    float ssum = 0.0f;
    for (int c0 = t16; c0 < 1024; c0 += 16) {
      float e = __expf(cur[scofs(r16, c0)] - mm);
      cur[scofs(r16, c0)] = e;
      ssum += e;
    }
    red[r16 * 16 + t16] = ssum;
    __syncthreads();
    if (t16 == 0) {
      float s2 = 0.0f;
#pragma unroll
      for (int i = 0; i < 16; ++i) s2 += red[r16 * 16 + i];
      rowstat[r16] = 1.0f / s2;
    }
    __syncthreads();
    const float inv = rowstat[r16];
    for (int c0 = t16; c0 < 1024; c0 += 16) {
      float pv = cur[scofs(r16, c0)] * inv;
      cur[scofs(r16, c0)] = pv;
      avg[r16 * 1024 + c0] += pv * 0.125f;
    }
    __syncthreads();

    // ---- AV: wave w reduces m in [w*128, (w+1)*128) ----
    v8f o0 = {}, o1 = {};
#pragma unroll
    for (int kk = 0; kk < 4; ++kk) {
      const int mb = wave * 128 + kk * 32;
      const int arow = lane & 15;
      const int kof = (lane < 16) ? 0 : 8;
      v16h ap;
#pragma unroll
      for (int i = 0; i < 8; ++i) {
        ap[i]     = (_Float16)cur[scofs(arow, mb + kof + i)];
        ap[8 + i] = (_Float16)cur[scofs(arow, mb + 16 + kof + i)];
      }
      v16h b0 = load_b_rows_f16(vthead + mb, N_SEQ);                      // dh 0..15
      v16h b1 = load_b_rows_f16(vthead + (size_t)16 * N_SEQ + mb, N_SEQ); // dh 16..31
      o0 = wmma_f16(ap, b0, o0);
      o1 = wmma_f16(ap, b1, o1);
    }
#pragma unroll
    for (int i = 0; i < 8; ++i) {
      int rr = i + rofs;
      part[(wave * 16 + rr) * 32 + cn]      = o0[i];
      part[(wave * 16 + rr) * 32 + 16 + cn] = o1[i];
    }
    __syncthreads();

    // ---- cross-wave reduce + write context (N, B, E) f32 ----
    for (int e = threadIdx.x; e < 16 * 32; e += 256) {
      int rr = e >> 5, dh = e & 31;
      float s = 0.0f;
#pragma unroll
      for (int wv = 0; wv < 8; ++wv) s += part[(wv * 16 + rr) * 32 + dh];
      ctx[(((size_t)(nbase + rr)) * BATCH + b) * EMB + h * DH + dh] = s;
    }
    __syncthreads();
  }

  // ---- attn_avg (B, N, N) ----
  for (int e = threadIdx.x; e < AVG_FLOATS; e += 256) {
    int rr = e >> 10, cc = e & 1023;
    avg_out[((size_t)b * N_SEQ + nbase + rr) * (size_t)N_SEQ + cc] = avg[e];
  }
}

// ---------------- kernel 3: output projection ----------------
__global__ void out_proj_kernel(const float* __restrict__ ctx,
                                const float* __restrict__ w,
                                const float* __restrict__ bias,
                                float* __restrict__ out) {
  const int wave = threadIdx.x >> 5;
  const int lane = threadIdx.x & 31;
  const int tile = blockIdx.x * 8 + wave;
  const int tiles_j = EMB / 16;                      // 16
  const int mbase = (tile / tiles_j) * 16;
  const int jbase = (tile % tiles_j) * 16;

  v8f acc = {};
#pragma unroll
  for (int kb = 0; kb < EMB; kb += 32) {
    v16h a = load_a_f32(ctx + (size_t)mbase * EMB + kb, EMB);
    v16h b = load_b_rows_f32(w + (size_t)jbase * EMB + kb, EMB);
    acc = wmma_f16(a, b, acc);
  }

  const int cn = lane & 15;
  const int rofs = (lane >= 16) ? 8 : 0;
  const float bj = bias[jbase + cn];
#pragma unroll
  for (int i = 0; i < 8; ++i) {
    int row = mbase + i + rofs;
    out[(size_t)row * EMB + jbase + cn] = acc[i] + bj;
  }
}

// ---------------- launcher ----------------
extern "C" void kernel_launch(void* const* d_in, const int* in_sizes, int n_in,
                              void* d_out, int out_size, void* d_ws, size_t ws_size,
                              hipStream_t stream) {
  (void)in_sizes; (void)n_in; (void)out_size; (void)ws_size;
  const float* value = (const float*)d_in[0];
  const float* pos   = (const float*)d_in[1];
  const float* w_in  = (const float*)d_in[2];
  const float* b_in  = (const float*)d_in[3];
  const float* w_out = (const float*)d_in[4];
  const float* b_out = (const float*)d_in[5];

  char* ws = (char*)d_ws;
  _Float16* qh = (_Float16*)ws;  ws += (size_t)NBH * N_SEQ * DH * 2;   // 4 MB
  _Float16* kh = (_Float16*)ws;  ws += (size_t)NBH * N_SEQ * DH * 2;   // 4 MB
  _Float16* vt = (_Float16*)ws;  ws += (size_t)NBH * DH * N_SEQ * 2;   // 4 MB
  float*    ctx = (float*)ws;                                          // 8 MB

  float* out_main = (float*)d_out;
  float* avg_out  = out_main + (size_t)N_SEQ * BATCH * EMB;            // 2,097,152 offset

  qkv_proj_kernel<<<(ROWS / 16) * (QKV_COLS / 16) / 8, 256, 0, stream>>>(value, w_in, b_in, qh, kh, vt);

  attn_kernel<<<(N_SEQ / 16) * BATCH, 256, SMEM_FLOATS * sizeof(float), stream>>>(
      qh, kh, vt, pos, ctx, avg_out);

  out_proj_kernel<<<(ROWS / 16) * (EMB / 16) / 8, 256, 0, stream>>>(ctx, w_out, b_out, out_main);
}
